// EnchantedViaDFGLayer_52295521796421
// MI455X (gfx1250) — matrix operand order repaired and checked by reading, not compiled, not run on hardware
//
#include <hip/hip_runtime.h>
#include <math.h>

// ---------------- problem constants (match reference) ----------------
#define BGRAPH 64
#define NNODE  1024
#define DIM    128
#define KSEL   4            // K+1 = 4 farthest neighbors kept
#define NMALL  (BGRAPH * NNODE)   // 65536

typedef __attribute__((ext_vector_type(16))) _Float16 v16h;
typedef __attribute__((ext_vector_type(8)))  _Float16 v8h;
typedef __attribute__((ext_vector_type(8)))  float    v8f;

// ---------------- WMMA helpers (CDNA5 wave32 layouts) ----------------
__device__ inline v8f wmma_f32_16x16x32(v16h a, v16h b, v8f c) {
  return __builtin_amdgcn_wmma_f32_16x16x32_f16(
      /*neg_a=*/false, a, /*neg_b=*/false, b,
      /*c_mod=*/(short)0, c, /*reuse_a=*/false, /*reuse_b=*/false);
}

// A fragment: 16x32 f16, source row-major, base = &A[r0][k0], ld in elems.
// lane<16: row=lane, K = 0..15; lane>=16: row=lane-16, K = 16..31 (contiguous 16h)
__device__ inline v16h load_a16(const _Float16* base, int ld, int lane) {
  int m  = lane & 15;
  int hi = lane >> 4;
  const _Float16* p = base + (size_t)m * ld + hi * 16;
  v8h lo = *(const v8h*)(p);
  v8h up = *(const v8h*)(p + 8);
  v16h r;
#pragma unroll
  for (int j = 0; j < 8; ++j) { r[j] = lo[j]; r[j + 8] = up[j]; }
  return r;
}

// B fragment: 32x16 f16 with B(k,n) = Bt[n][k]; base = &Bt[n0][0], ld elems.
// lane<16: col=lane, K=0..15 ; lane>=16: col=lane-16, K=16..31 (contiguous 16h)
__device__ inline v16h load_b16(const _Float16* base, int ld, int k0, int lane) {
  int n  = lane & 15;
  int hi = lane >> 4;
  return *(const v16h*)(base + (size_t)n * ld + k0 + hi * 16);
}

__device__ inline void atomicAddF(float* p, float v) { unsafeAtomicAdd(p, v); }

// ---------------- init: zero accumulators, deg = 1 (self loop) -------
__global__ void k_init(float* acc_f, float* acc_s, float* deg_f, float* deg_s) {
  int i = blockIdx.x * blockDim.x + threadIdx.x;
  if (i < NMALL * DIM) { acc_f[i] = 0.f; acc_s[i] = 0.f; }
  if (i < NMALL)       { deg_f[i] = 1.f; deg_s[i] = 1.f; }
}

// ---------------- normalize rows -> xn (f16), raw cast -> xh (f16) ---
__global__ void k_norm(const float* __restrict__ x, _Float16* xn, _Float16* xh) {
  int tid  = threadIdx.x;
  int lane = tid & 31;
  int u    = blockIdx.x * 8 + (tid >> 5);
  const float4 v = *(const float4*)(x + (size_t)u * DIM + lane * 4);
  float ss = v.x * v.x + v.y * v.y + v.z * v.z + v.w * v.w;
#pragma unroll
  for (int o = 16; o > 0; o >>= 1) ss += __shfl_xor(ss, o, 32);
  float inv = 1.f / fmaxf(sqrtf(ss), 1e-8f);
  size_t base = (size_t)u * DIM + lane * 4;
  xn[base + 0] = (_Float16)(v.x * inv); xn[base + 1] = (_Float16)(v.y * inv);
  xn[base + 2] = (_Float16)(v.z * inv); xn[base + 3] = (_Float16)(v.w * inv);
  xh[base + 0] = (_Float16)v.x; xh[base + 1] = (_Float16)v.y;
  xh[base + 2] = (_Float16)v.z; xh[base + 3] = (_Float16)v.w;
}

// ------- convert + transpose weights to f16, n-major (Bt[n][k]) ------
__global__ void k_wconv(const float* Wf, const float* Ws, const float* Wg,
                        _Float16* WtF, _Float16* WtS, _Float16* WtG) {
  int i = blockIdx.x * blockDim.x + threadIdx.x;   // 65536 threads
  if (i < 16384) {
    int k = i >> 7, n = i & 127;
    WtF[n * 128 + k] = (_Float16)Wf[i];
  } else if (i < 32768) {
    int j = i - 16384; int k = j >> 7, n = j & 127;
    WtS[n * 128 + k] = (_Float16)Ws[j];
  } else if (i < 65536) {
    int j = i - 32768; int k = j >> 7, n = j & 127;   // k in 0..255
    WtG[n * 256 + k] = (_Float16)Wg[j];
  }
}

// ---- similarity (WMMA) + top-4 farthest per row + deg_f atomics -----
__device__ inline void ins4(float v, int i, float* bv, int* bi) {
  if (v > bv[3]) {
    bv[3] = v; bi[3] = i;
#pragma unroll
    for (int j = 3; j > 0; --j) {
      if (bv[j] > bv[j - 1]) {
        float tv = bv[j]; bv[j] = bv[j - 1]; bv[j - 1] = tv;
        int   ti = bi[j]; bi[j] = bi[j - 1]; bi[j - 1] = ti;
      } else break;
    }
  }
}

__global__ void k_sim_topk(const _Float16* __restrict__ xn,
                           int* __restrict__ fidx, float* __restrict__ deg_f) {
  __shared__ float tile[16 * 512];       // 32 KB: 16 rows x 512 cols (one pass)
  __shared__ float cval[16][8][4];
  __shared__ int   cidx[16][8][4];

  int b  = blockIdx.x >> 6;              // graph
  int rb = blockIdx.x & 63;              // 16-row block within graph
  const _Float16* Xg = xn + (size_t)b * NNODE * DIM;
  int tid = threadIdx.x, wave = tid >> 5, lane = tid & 31;
  int hi = lane >> 4;

  // preload A fragments for the 16-row strip (K = 128 -> 4 tiles of 32)
  v16h a[4];
#pragma unroll
  for (int kk = 0; kk < 4; ++kk)
    a[kk] = load_a16(Xg + (size_t)(rb * 16) * DIM + kk * 32, DIM, lane);

  float bv[4] = {-1e30f, -1e30f, -1e30f, -1e30f};
  int   bi[4] = {0, 0, 0, 0};

  for (int pass = 0; pass < 2; ++pass) {
    // 512 columns: 32 col-tiles split across 8 waves
    for (int cb = wave; cb < 32; cb += 8) {
      int c0 = pass * 512 + cb * 16;
      v8f c = {};
#pragma unroll
      for (int kk = 0; kk < 4; ++kk) {
        v16h bf = load_b16(Xg + (size_t)c0 * DIM, DIM, kk * 32, lane);
        c = wmma_f32_16x16x32(a[kk], bf, c);
      }
      int n = lane & 15;
#pragma unroll
      for (int r = 0; r < 8; ++r) {
        int row = r + hi * 8;
        tile[row * 512 + cb * 16 + n] = 1.0f - c[r];   // cosine distance
      }
    }
    __syncthreads();
    if (tid < 128) {                       // 16 rows x 8 segments of 64 cols
      int row = tid >> 3, seg = tid & 7;
      const float* trow = tile + row * 512 + seg * 64;
#pragma unroll 4
      for (int j = 0; j < 64; ++j)
        ins4(trow[j], pass * 512 + seg * 64 + j, bv, bi);
    }
    __syncthreads();
  }

  if (tid < 128) {
    int row = tid >> 3, seg = tid & 7;
#pragma unroll
    for (int j = 0; j < 4; ++j) { cval[row][seg][j] = bv[j]; cidx[row][seg][j] = bi[j]; }
  }
  __syncthreads();

  if (tid < 16) {                          // merge 32 candidates -> top 4
    int row = tid;
    int g = b * NNODE + rb * 16 + row;
#pragma unroll
    for (int jj = 0; jj < 4; ++jj) {
      float best = -1e30f; int bseg = 0, bslot = 0;
      for (int s = 0; s < 8; ++s)
        for (int q = 0; q < 4; ++q)
          if (cval[row][s][q] > best) { best = cval[row][s][q]; bseg = s; bslot = q; }
      int dst = b * NNODE + cidx[row][bseg][bslot];
      cval[row][bseg][bslot] = -1e30f;     // exclude
      fidx[g * KSEL + jj] = dst;
      atomicAddF(&deg_f[dst], 1.0f);
    }
  }
}

// ---------------- structural degree histogram ------------------------
__global__ void k_degs(const int* __restrict__ ei, float* deg_s, int E) {
  int i = blockIdx.x * blockDim.x + threadIdx.x;
  if (i < E) atomicAddF(&deg_s[ei[E + i]], 1.0f);
}

// ---------------- deg -> rsqrt(deg) in place -------------------------
__global__ void k_dinv(float* deg) {
  int i = blockIdx.x * blockDim.x + threadIdx.x;
  if (i < NMALL) { float d = deg[i]; deg[i] = d > 0.f ? rsqrtf(d) : 0.f; }
}

// ---------------- h = X @ W  (WMMA, f16 in / f32 out) ----------------
__global__ void k_gemm_xw(const _Float16* __restrict__ A,
                          const _Float16* __restrict__ Bt,
                          float* __restrict__ C) {
  int r0 = blockIdx.x * 16;
  int tid = threadIdx.x, wave = tid >> 5, lane = tid & 31;
  int n0 = wave * 16;
  int hi = lane >> 4;
  v8f c = {};
#pragma unroll
  for (int kk = 0; kk < 4; ++kk) {
    v16h a = load_a16(A + (size_t)r0 * DIM + kk * 32, DIM, lane);
    v16h b = load_b16(Bt + (size_t)n0 * DIM, DIM, kk * 32, lane);
    c = wmma_f32_16x16x32(a, b, c);
  }
  int col = n0 + (lane & 15);
#pragma unroll
  for (int r = 0; r < 8; ++r)
    C[(size_t)(r0 + r + hi * 8) * DIM + col] = c[r];
}

// --------- feature-graph aggregation (4 edges per source node) -------
__global__ void k_feat_agg(const float* __restrict__ h, const int* __restrict__ fidx,
                           const float* __restrict__ dinv, float* acc) {
  int u = blockIdx.x, t = threadIdx.x;
  float du = dinv[u];
  float hv = h[(size_t)u * DIM + t];
#pragma unroll
  for (int j = 0; j < KSEL; ++j) {
    int v = fidx[u * KSEL + j];
    atomicAddF(&acc[(size_t)v * DIM + t], hv * du * dinv[v]);
  }
}

// --------- structural aggregation: one wave per edge, float4 ---------
__global__ void k_stru_agg(const float* __restrict__ h, const int* __restrict__ ei,
                           const float* __restrict__ dinv, float* acc, int E) {
  int tid = threadIdx.x;
  int e = blockIdx.x * 8 + (tid >> 5);
  if (e >= E) return;
  int lane = tid & 31;
  int s = ei[e], d = ei[E + e];
  float nrm = dinv[s] * dinv[d];
  const float4 hv = *(const float4*)(h + (size_t)s * DIM + lane * 4);
  float* p = acc + (size_t)d * DIM + lane * 4;
  atomicAddF(p + 0, hv.x * nrm);
  atomicAddF(p + 1, hv.y * nrm);
  atomicAddF(p + 2, hv.z * nrm);
  atomicAddF(p + 3, hv.w * nrm);
}

// ------------- block (128) sum reduction helper ----------------------
__device__ inline float block_sum128(float v, float* s, int t) {
  s[t] = v; __syncthreads();
  for (int o = 64; o > 0; o >>= 1) { if (t < o) s[t] += s[t + o]; __syncthreads(); }
  float r = s[0]; __syncthreads();
  return r;
}

// ---- branch post: + self loop + bias, relu, LayerNorm, f16 copy -----
__global__ void k_branch_post(float* __restrict__ acc, const float* __restrict__ h,
                              const float* __restrict__ dinv,
                              const float* __restrict__ bias,
                              const float* __restrict__ gam,
                              const float* __restrict__ bet,
                              _Float16* __restrict__ out_h) {
  __shared__ float red[128];
  int u = blockIdx.x, t = threadIdx.x;
  float du = dinv[u];
  size_t o = (size_t)u * DIM + t;
  float v = acc[o] + h[o] * du * du + bias[t];
  v = fmaxf(v, 0.f);
  float mean = block_sum128(v, red, t) * (1.f / DIM);
  float d = v - mean;
  float var = block_sum128(d * d, red, t) * (1.f / DIM);
  float y = d * rsqrtf(var + 1e-5f) * gam[t] + bet[t];
  acc[o] = y;                       // LayerNorm result, in place
  out_h[o] = (_Float16)y;           // f16 copy for gate GEMM
}

// ---- gate GEMM (K=256, WMMA) + sigmoid + gated fusion ---------------
__global__ void k_gate_fuse(const _Float16* __restrict__ featH,
                            const _Float16* __restrict__ struH,
                            const _Float16* __restrict__ WtG,
                            const float* __restrict__ b_gate,
                            const float* __restrict__ featF,
                            const float* __restrict__ struF,
                            float* __restrict__ fus) {
  int r0 = blockIdx.x * 16;
  int tid = threadIdx.x, wave = tid >> 5, lane = tid & 31;
  int n0 = wave * 16;
  int hi = lane >> 4;
  v8f c = {};
#pragma unroll
  for (int kt = 0; kt < 8; ++kt) {
    int k0 = kt * 32;
    const _Float16* Ap = (k0 < 128)
        ? featH + (size_t)r0 * DIM + k0
        : struH + (size_t)r0 * DIM + (k0 - 128);
    v16h a = load_a16(Ap, DIM, lane);
    v16h b = load_b16(WtG + (size_t)n0 * 256, 256, k0, lane);
    c = wmma_f32_16x16x32(a, b, c);
  }
  int col = n0 + (lane & 15);
#pragma unroll
  for (int r = 0; r < 8; ++r) {
    int row = r0 + r + hi * 8;
    size_t o = (size_t)row * DIM + col;
    float z = c[r] + b_gate[col];
    float g = 1.f / (1.f + expf(-z));
    fus[o] = g * featF[o] + (1.f - g) * struF[o];
  }
}

// ---- final LayerNorm + residual -------------------------------------
__global__ void k_final_ln(const float* __restrict__ fus, const float* __restrict__ x,
                           const float* __restrict__ gam, const float* __restrict__ bet,
                           float* __restrict__ out) {
  __shared__ float red[128];
  int u = blockIdx.x, t = threadIdx.x;
  size_t o = (size_t)u * DIM + t;
  float v = fus[o];
  float mean = block_sum128(v, red, t) * (1.f / DIM);
  float d = v - mean;
  float var = block_sum128(d * d, red, t) * (1.f / DIM);
  out[o] = d * rsqrtf(var + 1e-5f) * gam[t] + bet[t] + x[o];
}

// =====================================================================
extern "C" void kernel_launch(void* const* d_in, const int* in_sizes, int n_in,
                              void* d_out, int out_size, void* d_ws, size_t ws_size,
                              hipStream_t stream) {
  const float* x      = (const float*)d_in[0];
  const int*   ei     = (const int*)d_in[1];
  const float* W_feat = (const float*)d_in[2];
  const float* b_feat = (const float*)d_in[3];
  const float* W_stru = (const float*)d_in[4];
  const float* b_stru = (const float*)d_in[5];
  const float* W_gate = (const float*)d_in[6];
  const float* b_gate = (const float*)d_in[7];
  const float* g_feat = (const float*)d_in[8];
  const float* be_feat= (const float*)d_in[9];
  const float* g_stru = (const float*)d_in[10];
  const float* be_stru= (const float*)d_in[11];
  const float* g_fus  = (const float*)d_in[12];
  const float* be_fus = (const float*)d_in[13];
  float* out = (float*)d_out;
  const int E = in_sizes[1] / 2;

  // ---- workspace carve ----
  size_t off = 0;
  auto carve = [&](size_t bytes) -> void* {
    void* p = (char*)d_ws + off;
    off += (bytes + 255) & ~(size_t)255;
    return p;
  };
  _Float16* xn    = (_Float16*)carve((size_t)NMALL * DIM * 2);
  _Float16* xh    = (_Float16*)carve((size_t)NMALL * DIM * 2);
  float* h_feat   = (float*)carve((size_t)NMALL * DIM * 4);   // later reused as `fus`
  float* h_stru   = (float*)carve((size_t)NMALL * DIM * 4);
  float* acc_feat = (float*)carve((size_t)NMALL * DIM * 4);
  float* acc_stru = (float*)carve((size_t)NMALL * DIM * 4);
  _Float16* featH = (_Float16*)carve((size_t)NMALL * DIM * 2);
  _Float16* struH = (_Float16*)carve((size_t)NMALL * DIM * 2);
  int*   fidx     = (int*)carve((size_t)NMALL * KSEL * 4);
  float* deg_f    = (float*)carve((size_t)NMALL * 4);
  float* deg_s    = (float*)carve((size_t)NMALL * 4);
  _Float16* WtF   = (_Float16*)carve(128 * 128 * 2);
  _Float16* WtS   = (_Float16*)carve(128 * 128 * 2);
  _Float16* WtG   = (_Float16*)carve(128 * 256 * 2);
  (void)ws_size; (void)n_in; (void)out_size;

  // 1) init accumulators + self-loop degree
  k_init<<<(NMALL * DIM + 255) / 256, 256, 0, stream>>>(acc_feat, acc_stru, deg_f, deg_s);
  // 2) row-normalize + f16 casts
  k_norm<<<NMALL / 8, 256, 0, stream>>>(x, xn, xh);
  // 3) weight convert + transpose
  k_wconv<<<256, 256, 0, stream>>>(W_feat, W_stru, W_gate, WtF, WtS, WtG);
  // 4) WMMA cosine similarity + top-4 farthest + deg_f
  k_sim_topk<<<BGRAPH * (NNODE / 16), 256, 0, stream>>>(xn, fidx, deg_f);
  // 5) structural degrees
  k_degs<<<(E + 255) / 256, 256, 0, stream>>>(ei, deg_s, E);
  // 6) deg -> rsqrt
  k_dinv<<<(NMALL + 255) / 256, 256, 0, stream>>>(deg_f);
  k_dinv<<<(NMALL + 255) / 256, 256, 0, stream>>>(deg_s);
  // 7) dense GEMMs h = x @ W  (WMMA)
  k_gemm_xw<<<NMALL / 16, 256, 0, stream>>>(xh, WtF, h_feat);
  k_gemm_xw<<<NMALL / 16, 256, 0, stream>>>(xh, WtS, h_stru);
  // 8) aggregations (normalized scatter-add)
  k_feat_agg<<<NMALL, 128, 0, stream>>>(h_feat, fidx, deg_f, acc_feat);
  k_stru_agg<<<(E + 7) / 8, 256, 0, stream>>>(h_stru, ei, deg_s, acc_stru, E);
  // 9) self-loop + bias + relu + LayerNorm per branch (in place) + f16 copy
  k_branch_post<<<NMALL, 128, 0, stream>>>(acc_feat, h_feat, deg_f, b_feat, g_feat, be_feat, featH);
  k_branch_post<<<NMALL, 128, 0, stream>>>(acc_stru, h_stru, deg_s, b_stru, g_stru, be_stru, struH);
  // 10) gate GEMM (WMMA, K=256) + sigmoid + fusion  (reuse h_feat as fus)
  k_gate_fuse<<<NMALL / 16, 256, 0, stream>>>(featH, struH, WtG, b_gate, acc_feat, acc_stru, h_feat);
  // 11) final LayerNorm + residual
  k_final_ln<<<NMALL, 128, 0, stream>>>(h_feat, x, g_fus, be_fus, out);
}
